// AttentionModel_39848706572730
// MI455X (gfx1250) — matrix operand order repaired and checked by reading, not compiled, not run on hardware
//
#include <hip/hip_runtime.h>
#include <hip/hip_bf16.h>

// CDNA5 / gfx1250 attention pipeline, bf16 WMMA (v_wmma_f32_16x16x32_bf16)
// with double-buffered Tensor-Data-Mover staging of K tiles into LDS
// (tensor_load_to_lds + s_wait_tensorcnt ping-pong, padded LDS layout).
//
//  K0a: Wout f32 -> bf16 (ws)
//  K0b: K    f32 -> bf16 (ws)       -- enables pure-bf16 TDM staging in K1
//  K0c: V    f32 -> bf16 transposed Vt[B][E][S] (ws) -- K2 goes LDS-free
//  K1 : two-pass online-softmax: QK^T (WMMA) -> stats, then QK^T again ->
//       probs = exp(s/32 - m)/l * dropout_mask, stored bf16 in ws
//  K2 : attn = probs @ V (WMMA, direct global fragments from Vt), bf16 in ws
//  K3 : out  = attn @ Wout^T + bout (WMMA, direct global fragments)

typedef __attribute__((ext_vector_type(16))) __bf16          v16bf;
typedef __attribute__((ext_vector_type(8)))  float           v8f;
typedef __attribute__((ext_vector_type(8)))  unsigned short  u16x8;
typedef __attribute__((ext_vector_type(16))) unsigned short  u16x16;
typedef __attribute__((ext_vector_type(4)))  unsigned int    u32x4;
typedef __attribute__((ext_vector_type(8)))  int             i32x8;
typedef __attribute__((ext_vector_type(4)))  int             i32x4;

#define B_ 8
#define S_ 2048
#define E_ 1024
#define QSTR (E_ + 8)   // LDS stride (halves) for Q tile: 2064B = 129*16
#define KSTR (E_ + 32)  // LDS stride (halves) for TDM K tiles: 1024 + 2x16 pad
#define VTT 72          // LDS stride (halves) in transpose kernel tile
#define KT_OFF0 (64 * QSTR * 2)             // byte offset of K buffer 0
#define KT_OFF1 (KT_OFF0 + 32 * KSTR * 2)   // byte offset of K buffer 1

// TDM availability: device pass only; arity differs between clang-22 (ROCm
// 7.2, 5 args) and clang-23 (amdgpu-toolchain, 6 args).
#if defined(__HIP_DEVICE_COMPILE__) && __has_builtin(__builtin_amdgcn_tensor_load_to_lds)
#define HAVE_TDM 1
#else
#define HAVE_TDM 0
#endif

#define WMMA_BF16(A, Bv, C) \
  __builtin_amdgcn_wmma_f32_16x16x32_bf16(false, (A), false, (Bv), (short)0, (C), false, false)

__device__ __forceinline__ unsigned short f2bf(float f) {
  unsigned int u = __builtin_bit_cast(unsigned int, f);
  u += 0x7FFFu + ((u >> 16) & 1u);  // round-to-nearest-even
  return (unsigned short)(u >> 16);
}

// Build a v16bf fragment from two 16-byte aligned groups of 8 bf16 values.
__device__ __forceinline__ v16bf make_frag(const unsigned short* p0,
                                           const unsigned short* p1) {
  u16x8 lo = *(const u16x8*)p0;
  u16x8 hi = *(const u16x8*)p1;
  u16x16 u = __builtin_shufflevector(lo, hi, 0, 1, 2, 3, 4, 5, 6, 7,
                                     8, 9, 10, 11, 12, 13, 14, 15);
  return __builtin_bit_cast(v16bf, u);
}

// Reductions across the 16 lanes holding one C-matrix row (xor<16 stays in half).
__device__ __forceinline__ float rowmax16(float v) {
  v = fmaxf(v, __shfl_xor(v, 1, 32));
  v = fmaxf(v, __shfl_xor(v, 2, 32));
  v = fmaxf(v, __shfl_xor(v, 4, 32));
  v = fmaxf(v, __shfl_xor(v, 8, 32));
  return v;
}
__device__ __forceinline__ float rowsum16(float v) {
  v += __shfl_xor(v, 1, 32);
  v += __shfl_xor(v, 2, 32);
  v += __shfl_xor(v, 4, 32);
  v += __shfl_xor(v, 8, 32);
  return v;
}

// Stage a 64 x 1024 f32 tile from global into LDS as bf16 (row stride QSTR).
__device__ __forceinline__ void stage_tile_f32_bf16(unsigned short* dst,
                                                    const float* __restrict__ src,
                                                    int tid) {
#pragma unroll 4
  for (int it = 0; it < 64; ++it) {
    int idx = tid + 256 * it;  // 16384 float4 total
    int row = idx >> 8;        // 256 float4 per row
    int c4  = idx & 255;
    float4 v = *(const float4*)(src + (size_t)row * E_ + (size_t)c4 * 4);
    unsigned int lo = (unsigned int)f2bf(v.x) | ((unsigned int)f2bf(v.y) << 16);
    unsigned int hi = (unsigned int)f2bf(v.z) | ((unsigned int)f2bf(v.w) << 16);
    *(uint2*)(dst + (size_t)row * QSTR + c4 * 4) = make_uint2(lo, hi);
  }
}

// Fallback manual 32-key tile copy (bf16 source) producing the exact TDM
// padded layout: LDS half-offset = row*KSTR + e + 16*(e>>9).
__device__ __forceinline__ void stage_ktile32(unsigned short* dst,
                                              const unsigned short* __restrict__ src,
                                              int tid) {
#pragma unroll 4
  for (int it = 0; it < 16; ++it) {
    int idx = tid + 256 * it;  // 4096 x uint4 (8 halves each)
    int row = idx >> 7;        // 128 uint4 per row
    int e   = (idx & 127) * 8;
    uint4 v = *(const uint4*)(src + (size_t)row * E_ + e);
    *(uint4*)(dst + (size_t)row * KSTR + e + ((e >> 9) << 4)) = v;
  }
}

#if HAVE_TDM
// Issue one TDM 2D tile load: 32 rows x 1024 bf16 from gsrc into LDS at
// lds_off, padding 8 DWORDs after every 256 DWORDs (-> KSTR layout).
__device__ __forceinline__ void tdm_load_ktile32(unsigned int lds_off,
                                                 const unsigned short* gsrc) {
  unsigned long long ga = (unsigned long long)(size_t)gsrc;
  // D# group0: count=1 | lds_addr | global_addr[56:0] | type=2
  u32x4 g0 = {1u, lds_off, (unsigned)(ga & 0xFFFFFFFFu),
              (unsigned)(((ga >> 32) & 0x01FFFFFFu) | 0x80000000u)};
  // D# group1:
  //  w0: data_size=1(2B)<<16 | pad_enable<<20 | pad_interval=7(256dw)<<22
  //      | pad_amount=7(8dw)<<25                     = 0x0FD10000
  //  w1[31:16]=tensor_dim0.lo=1024, w2[31:16]=tensor_dim1.lo=32,
  //  w3[31:16]=tile_dim0=1024, w4[15:0]=tile_dim1=32,
  //  w5=tensor_dim0_stride.lo=1024
  i32x8 g1 = {(int)0x0FD10000, (int)0x04000000, (int)0x00200000,
              (int)0x04000000, 0x20, 0x400, 0, 0};
  i32x4 z4 = {0, 0, 0, 0};
#if __clang_major__ >= 23
  i32x8 z8 = {0, 0, 0, 0, 0, 0, 0, 0};
  __builtin_amdgcn_tensor_load_to_lds(g0, g1, z4, z4, z8, 0);
#else
  __builtin_amdgcn_tensor_load_to_lds(g0, g1, z4, z4, 0);
#endif
}
#endif

// One 16x16 score tile: rows 16*rb..+15, keys 16*kh..+15 of the current
// 32-key tile, accumulated over the full E=1024 inner dimension.
__device__ __forceinline__ void score_tile32(const unsigned short* Qt,
                                             const unsigned short* Kt,
                                             int rb, int kh, int lane, v8f* o) {
  v8f c = {};
  const unsigned short* arow = Qt + (size_t)(16 * rb + (lane & 15)) * QSTR;
  const unsigned short* brow = Kt + (size_t)(16 * kh + (lane & 15)) * KSTR;
  const int koffA = (lane & 16) ? 16 : 0;  // A: lanes>=16 hold K 16-31
  const int koffB = (lane & 16) ? 16 : 0;  // B: lanes>=16 hold K 16-31
#pragma unroll 4
  for (int kc = 0; kc < E_ / 32; ++kc) {
    const int abase = kc * 32 + koffA;
    const int e     = kc * 32 + koffB;
    const int boff  = e + ((e >> 9) << 4);  // TDM pad bump (uniform per frag)
    v16bf a  = make_frag(arow + abase, arow + abase + 8);
    v16bf bv = make_frag(brow + boff, brow + boff + 8);
    c = WMMA_BF16(a, bv, c);
  }
  *o = c;
}

// ---------------- K0a/K0b: f32 -> bf16 streaming convert ----------------
__global__ __launch_bounds__(256) void cvt_f32_bf16_kernel(
    const float* __restrict__ src, unsigned short* __restrict__ dst) {
  size_t i = (size_t)blockIdx.x * 256 + threadIdx.x;  // float4 index
  float4 v = ((const float4*)src)[i];
  unsigned int lo = (unsigned int)f2bf(v.x) | ((unsigned int)f2bf(v.y) << 16);
  unsigned int hi = (unsigned int)f2bf(v.z) | ((unsigned int)f2bf(v.w) << 16);
  *(uint2*)(dst + 4 * i) = make_uint2(lo, hi);
}

// ---------------- K0c: V f32 -> bf16, transposed to Vt[B][E][S] ----------------
__global__ __launch_bounds__(256) void vt_kernel(const float* __restrict__ V,
                                                 unsigned short* __restrict__ Vt) {
  __shared__ __align__(16) unsigned short tile[64 * VTT];  // [e'][s'] bf16
  const int b   = blockIdx.y;
  const int s0  = (blockIdx.x >> 4) * 64;  // 32 s-tiles
  const int e0  = (blockIdx.x & 15) * 64;  // 16 e-tiles
  const int tid = threadIdx.x;
#pragma unroll
  for (int it = 0; it < 4; ++it) {
    int idx = tid + 256 * it;  // 1024 float4 (64 s-rows x 16 float4)
    int row = idx >> 4;
    int c4  = idx & 15;
    float4 v = *(const float4*)(V + ((size_t)b * S_ + s0 + row) * E_ + e0 + c4 * 4);
    int e = c4 * 4;
    tile[(e + 0) * VTT + row] = f2bf(v.x);
    tile[(e + 1) * VTT + row] = f2bf(v.y);
    tile[(e + 2) * VTT + row] = f2bf(v.z);
    tile[(e + 3) * VTT + row] = f2bf(v.w);
  }
  __syncthreads();
#pragma unroll
  for (int it = 0; it < 2; ++it) {
    int idx  = tid + 256 * it;  // 512 uint4 (64 e-rows x 8 uint4)
    int erow = idx >> 3;
    int c8   = (idx & 7) * 8;
    uint4 pk = *(const uint4*)&tile[erow * VTT + c8];
    *(uint4*)(Vt + ((size_t)b * E_ + e0 + erow) * S_ + s0 + c8) = pk;
  }
}

// ---------------- K1: softmax stats + probs ----------------
__global__ __launch_bounds__(256) void attn_probs_kernel(
    const float* __restrict__ Q, const unsigned short* __restrict__ Kbf,
    const float* __restrict__ mask, const int* __restrict__ inv_scale,
    unsigned short* __restrict__ probs) {
  extern __shared__ unsigned short smem[];
  unsigned short* Qt  = smem;                  // [64][QSTR] bf16
  unsigned short* Kt0 = Qt + 64 * QSTR;        // [32][KSTR] bf16, ping
  unsigned short* Kt1 = Kt0 + 32 * KSTR;       // [32][KSTR] bf16, pong
  float* statsM = (float*)(Kt1 + 32 * KSTR);   // [2][4][2][8]
  float* statsL = statsM + 128;

  const int b    = blockIdx.y;
  const int q0   = blockIdx.x * 64;
  const int tid  = threadIdx.x;
  const int lane = tid & 31;
  const int wave = tid >> 5;
  const int rb   = wave >> 1;  // row block (16 rows)
  const int kh   = wave & 1;   // key half (16 keys of the 32-key tile)
  const int half = lane >> 4;  // C layout: lanes>=16 hold M = r+8
  const float scale = 1.0f / (float)(*inv_scale);
  const unsigned short* ksrc = Kbf + (size_t)b * S_ * E_;

  stage_tile_f32_bf16(Qt, Q + ((size_t)b * S_ + q0) * E_, tid);

  float mrun[8], lrun[8];
#pragma unroll
  for (int r = 0; r < 8; ++r) { mrun[r] = -__builtin_inff(); lrun[r] = 0.f; }

  // ---- pass 1: online row max / sum-exp, double-buffered K staging ----
#if HAVE_TDM
  if (wave == 0) {
    tdm_load_ktile32(KT_OFF0, ksrc);
    __builtin_amdgcn_s_wait_tensorcnt(0);
  }
#else
  stage_ktile32(Kt0, ksrc, tid);
#endif
  __syncthreads();
  for (int kt = 0; kt < S_ / 32; ++kt) {
    unsigned short* cur = (kt & 1) ? Kt1 : Kt0;
#if HAVE_TDM
    if (wave == 0 && kt + 1 < S_ / 32)
      tdm_load_ktile32((kt & 1) ? KT_OFF0 : KT_OFF1,
                       ksrc + (size_t)(kt + 1) * 32 * E_);
#else
    unsigned short* nxt = (kt & 1) ? Kt0 : Kt1;
    if (kt + 1 < S_ / 32) stage_ktile32(nxt, ksrc + (size_t)(kt + 1) * 32 * E_, tid);
#endif
    v8f c;
    score_tile32(Qt, cur, rb, kh, lane, &c);
#pragma unroll
    for (int r = 0; r < 8; ++r) {
      float s  = c[r] * scale;
      float tm = rowmax16(s);
      float ts = rowsum16(__expf(s - tm));
      float nm = fmaxf(mrun[r], tm);
      lrun[r] = lrun[r] * __expf(mrun[r] - nm) + ts * __expf(tm - nm);
      mrun[r] = nm;
    }
#if HAVE_TDM
    if (wave == 0) __builtin_amdgcn_s_wait_tensorcnt(0);
#endif
    __syncthreads();
  }

  // ---- merge the two key-half waves' stats through LDS ----
  if ((lane & 15) == 0) {
#pragma unroll
    for (int r = 0; r < 8; ++r) {
      int idx = ((kh * 4 + rb) * 2 + half) * 8 + r;
      statsM[idx] = mrun[r];
      statsL[idx] = lrun[r];
    }
  }
  __syncthreads();
  float mfin[8], linv[8];
#pragma unroll
  for (int r = 0; r < 8; ++r) {
    int i0 = ((0 * 4 + rb) * 2 + half) * 8 + r;
    int i1 = ((1 * 4 + rb) * 2 + half) * 8 + r;
    float m0 = statsM[i0], l0 = statsL[i0];
    float m1 = statsM[i1], l1 = statsL[i1];
    float nm = fmaxf(m0, m1);
    float l  = l0 * __expf(m0 - nm) + l1 * __expf(m1 - nm);
    mfin[r] = nm;
    linv[r] = 1.0f / l;
  }
  __syncthreads();

  // ---- pass 2: recompute scores, normalize, apply dropout, emit bf16 ----
#if HAVE_TDM
  if (wave == 0) {
    tdm_load_ktile32(KT_OFF0, ksrc);
    __builtin_amdgcn_s_wait_tensorcnt(0);
  }
#else
  stage_ktile32(Kt0, ksrc, tid);
#endif
  __syncthreads();
  for (int kt = 0; kt < S_ / 32; ++kt) {
    unsigned short* cur = (kt & 1) ? Kt1 : Kt0;
#if HAVE_TDM
    if (wave == 0 && kt + 1 < S_ / 32)
      tdm_load_ktile32((kt & 1) ? KT_OFF0 : KT_OFF1,
                       ksrc + (size_t)(kt + 1) * 32 * E_);
#else
    unsigned short* nxt = (kt & 1) ? Kt0 : Kt1;
    if (kt + 1 < S_ / 32) stage_ktile32(nxt, ksrc + (size_t)(kt + 1) * 32 * E_, tid);
#endif
    v8f c;
    score_tile32(Qt, cur, rb, kh, lane, &c);
    const int key = kt * 32 + 16 * kh + (lane & 15);
#pragma unroll
    for (int r = 0; r < 8; ++r) {
      int qr = q0 + 16 * rb + r + 8 * half;
      size_t rowbase = ((size_t)b * S_ + qr) * (size_t)S_;
      float p = __expf(c[r] * scale - mfin[r]) * linv[r] * mask[rowbase + key];
      probs[rowbase + key] = f2bf(p);
    }
#if HAVE_TDM
    if (wave == 0) __builtin_amdgcn_s_wait_tensorcnt(0);
#endif
    __syncthreads();
  }
}

// ---------------- K2: attn = probs @ V (Vt pre-transposed, LDS-free) ------
__global__ __launch_bounds__(256) void pv_kernel(const unsigned short* __restrict__ probs,
                                                 const unsigned short* __restrict__ VtG,
                                                 unsigned short* __restrict__ attn) {
  const int b    = blockIdx.y;
  const int q0   = (blockIdx.x >> 2) * 64;
  const int e0   = (blockIdx.x & 3) * 256;
  const int tid  = threadIdx.x;
  const int lane = tid & 31;
  const int wave = tid >> 5;
  const int rb   = wave & 3;   // 4 row blocks of 16
  const int cp   = wave >> 2;  // 2 column halves of 128
  const int half = lane >> 4;
  const int koffA = (lane & 16) ? 16 : 0;
  const int koffB = (lane & 16) ? 16 : 0;

  v8f acc[8] = {};
  const unsigned short* arow =
      probs + ((size_t)b * S_ + q0 + 16 * rb + (lane & 15)) * (size_t)S_;
  const unsigned short* brow0 =
      VtG + ((size_t)b * E_ + e0 + 128 * cp + (lane & 15)) * (size_t)S_;

#pragma unroll 2
  for (int kc = 0; kc < S_ / 32; ++kc) {
    const int kb = kc * 32;
    if (kc + 1 < S_ / 32) __builtin_prefetch(arow + kb + 32, 0, 0);
    v16bf a = make_frag(arow + kb + koffA, arow + kb + koffA + 8);
#pragma unroll
    for (int i = 0; i < 8; ++i) {
      const unsigned short* brow = brow0 + (size_t)(16 * i) * S_ + kb + koffB;
      v16bf bv = make_frag(brow, brow + 8);
      acc[i] = WMMA_BF16(a, bv, acc[i]);
    }
  }
#pragma unroll
  for (int i = 0; i < 8; ++i) {
#pragma unroll
    for (int r = 0; r < 8; ++r) {
      int qr = q0 + 16 * rb + r + 8 * half;
      int e  = e0 + 128 * cp + 16 * i + (lane & 15);
      attn[((size_t)b * S_ + qr) * (size_t)E_ + e] = f2bf(acc[i][r]);
    }
  }
}

// ---------------- K3: out = attn @ Wout^T + bout ----------------
__global__ __launch_bounds__(256) void out_proj_kernel(
    const unsigned short* __restrict__ attn, const unsigned short* __restrict__ Wb,
    const float* __restrict__ bout, float* __restrict__ out) {
  const int b    = blockIdx.y;
  const int q0   = (blockIdx.x >> 2) * 64;
  const int e0   = (blockIdx.x & 3) * 256;
  const int tid  = threadIdx.x;
  const int lane = tid & 31;
  const int wave = tid >> 5;
  const int rb   = wave & 3;
  const int cp   = wave >> 2;
  const int half = lane >> 4;
  const int koffA = (lane & 16) ? 16 : 0;
  const int koffB = (lane & 16) ? 16 : 0;

  v8f acc[8] = {};
  const unsigned short* arow =
      attn + ((size_t)b * S_ + q0 + 16 * rb + (lane & 15)) * (size_t)E_;
  const unsigned short* brow0 =
      Wb + (size_t)(e0 + 128 * cp + (lane & 15)) * (size_t)E_;  // Wout row = B col

#pragma unroll 2
  for (int kc = 0; kc < E_ / 32; ++kc) {
    const int kb = kc * 32;
    if (kc + 1 < E_ / 32) __builtin_prefetch(arow + kb + 32, 0, 0);
    v16bf a = make_frag(arow + kb + koffA, arow + kb + koffA + 8);
#pragma unroll
    for (int i = 0; i < 8; ++i) {
      const unsigned short* brow = brow0 + (size_t)(16 * i) * E_ + kb + koffB;
      v16bf bv = make_frag(brow, brow + 8);
      acc[i] = WMMA_BF16(a, bv, acc[i]);
    }
  }
#pragma unroll
  for (int i = 0; i < 8; ++i) {
#pragma unroll
    for (int r = 0; r < 8; ++r) {
      int qr = q0 + 16 * rb + r + 8 * half;
      int e  = e0 + 128 * cp + 16 * i + (lane & 15);
      out[((size_t)b * S_ + qr) * (size_t)E_ + e] = acc[i][r] + bout[e];
    }
  }
}

extern "C" void kernel_launch(void* const* d_in, const int* in_sizes, int n_in,
                              void* d_out, int out_size, void* d_ws, size_t ws_size,
                              hipStream_t stream) {
  (void)in_sizes; (void)n_in; (void)out_size; (void)ws_size;
  const float* Q    = (const float*)d_in[0];
  const float* K    = (const float*)d_in[1];
  const float* V    = (const float*)d_in[2];
  const float* mask = (const float*)d_in[3];
  const float* Wout = (const float*)d_in[4];
  const float* bout = (const float*)d_in[5];
  const int* inv_scale = (const int*)d_in[6];
  float* out = (float*)d_out;

  // ws (bf16): probs[B,S,S] | attn[B,S,E] | Wout[E,E] | K[B,S,E] | Vt[B,E,S]
  unsigned short* probs = (unsigned short*)d_ws;
  unsigned short* attn  = probs + (size_t)B_ * S_ * S_;
  unsigned short* Wb    = attn + (size_t)B_ * S_ * E_;
  unsigned short* Kbf   = Wb + (size_t)E_ * E_;
  unsigned short* Vt    = Kbf + (size_t)B_ * S_ * E_;

  cvt_f32_bf16_kernel<<<(E_ * E_ / 4) / 256, 256, 0, stream>>>(Wout, Wb);
  cvt_f32_bf16_kernel<<<((size_t)B_ * S_ * E_ / 4) / 256, 256, 0, stream>>>(K, Kbf);
  vt_kernel<<<dim3((S_ / 64) * (E_ / 64), B_), 256, 0, stream>>>(V, Vt);

  size_t lds1 = (size_t)64 * QSTR * 2 + (size_t)64 * KSTR * 2 + 256 * sizeof(float);
  attn_probs_kernel<<<dim3(S_ / 64, B_), 256, lds1, stream>>>(Q, Kbf, mask, inv_scale, probs);

  pv_kernel<<<dim3((S_ / 64) * (E_ / 256), B_), 256, 0, stream>>>(probs, Vt, attn);

  out_proj_kernel<<<dim3((S_ / 64) * (E_ / 256), B_), 256, 0, stream>>>(attn, Wb, bout, out);
}
